// MultiDilatelocalAttention_6880537608924
// MI455X (gfx1250) — matrix operand order repaired and checked by reading, not compiled, not run on hardware
//
#include <hip/hip_runtime.h>

typedef __attribute__((ext_vector_type(16))) _Float16 v16h;
typedef __attribute__((ext_vector_type(8)))  _Float16 v8h;
typedef __attribute__((ext_vector_type(4)))  _Float16 v4h;
typedef __attribute__((ext_vector_type(8)))  float    v8f;

#define N_H    56
#define N_W    56
#define NPOS   3136            // 56*56
#define NBATCH 16
#define NROWS  (NBATCH * NPOS) // 50176
#define CIN    384
#define QKVO   1152
#define HDIM   32
#define BSTRIDE 392            // padded LDS row stride (halfs): 784B = 196 dw, 196%64=4

// ---------------------------------------------------------------------------
// CDNA5 async copy: global -> LDS, 16B per lane, tracked by ASYNCcnt.
// LDS address = low 32 bits of the generic (flat) address (LDS aperture rule).
// ---------------------------------------------------------------------------
__device__ __forceinline__ void async_copy_b128(void* lds_ptr, const void* gptr) {
    unsigned lds_off = (unsigned)(uintptr_t)lds_ptr;
    unsigned long long ga = (unsigned long long)(uintptr_t)gptr;
    asm volatile("global_load_async_to_lds_b128 %0, %1, off"
                 :: "v"(lds_off), "v"(ga) : "memory");
}
__device__ __forceinline__ void async_wait0() {
    asm volatile("s_wait_asynccnt 0" ::: "memory");
}

// ---------------------------------------------------------------------------
// Fragment loaders matching CDNA5 wave32 WMMA VGPR layouts (ISA 7.12.2).
// A (f16, 16x32): lane L holds row M=L&15; elems 0..7 -> K=kb+hsel*8+e,
//                 elems 8..15 -> K=kb+16+hsel*8+(e-8).  p = &A[row][kb+hsel*8].
// B (f16, 32x16): lane L holds col N=L&15; same K mapping as A.
//                 p = &B[col][kb+hsel*8].
// ---------------------------------------------------------------------------
__device__ __forceinline__ v16h load_a_frag(const _Float16* __restrict__ p) {
    v8h lo = *(const v8h*)(p);
    v8h hi = *(const v8h*)(p + 16);
    v16h r;
#pragma unroll
    for (int i = 0; i < 8; ++i) { r[i] = lo[i]; r[i + 8] = hi[i]; }
    return r;
}

__device__ __forceinline__ v16h load_b_frag_lds(const _Float16* p) {
    v8h lo = *(const v8h*)(p);
    v8h hi = *(const v8h*)(p + 16);
    v16h r;
#pragma unroll
    for (int i = 0; i < 8; ++i) { r[i] = lo[i]; r[i + 8] = hi[i]; }
    return r;
}

// ---------------------------------------------------------------------------
// Kernel 0: fp32 -> f16 conversion (x, w_qkv, w_proj), 4 elements per thread.
// ---------------------------------------------------------------------------
__global__ __launch_bounds__(256) void cvt_f32_f16_kernel(
        const float* __restrict__ src, _Float16* __restrict__ dst, int n4) {
    const int t = blockIdx.x * blockDim.x + threadIdx.x;
    if (t >= n4) return;
    float4 v = *(const float4*)(src + (size_t)t * 4);
    v4h h;
    h[0] = (_Float16)v.x; h[1] = (_Float16)v.y;
    h[2] = (_Float16)v.z; h[3] = (_Float16)v.w;
    *(v4h*)(dst + (size_t)t * 4) = h;
}

// ---------------------------------------------------------------------------
// Shared GEMM core: block = 128 rows x 128 cols; weight panel (128 cols x
// 384 K, f16) async-staged into LDS once; each of the 8 waves owns one 16-row
// tile and 8 WMMA accumulators over the 128 cols.
//   wsrc row layout: W[col][k], k contiguous (row-major [cols, 384]).
// ---------------------------------------------------------------------------
__device__ __forceinline__ void gemm_panel(
        const _Float16* __restrict__ ah, const _Float16* __restrict__ wsrc,
        _Float16* Bsh, int m_base, int n_base, int lane15, int hsel, v8f acc[8]) {
    // cooperative async fill: 128 rows x 48 16B-chunks = 6144 chunks
    const int tid = threadIdx.x;
#pragma unroll
    for (int i = 0; i < 24; ++i) {
        const int chunk = i * 256 + tid;
        const int col = chunk / 48;
        const int cw  = chunk % 48;
        async_copy_b128((char*)Bsh + col * (BSTRIDE * 2) + cw * 16,
                        (const char*)(wsrc + (size_t)n_base * CIN) + col * 768 + cw * 16);
    }
    async_wait0();
    __syncthreads();

    const int a_row = m_base + lane15;
    for (int kb = 0; kb < CIN; kb += 32) {
        v16h a = load_a_frag(ah + (size_t)a_row * CIN + kb + hsel * 8);
        const _Float16* bp = Bsh + (size_t)lane15 * BSTRIDE + kb + hsel * 8;
#pragma unroll
        for (int t = 0; t < 8; ++t) {
            v16h b = load_b_frag_lds(bp + t * 16 * BSTRIDE);
            acc[t] = __builtin_amdgcn_wmma_f32_16x16x32_f16(
                         false, a, false, b, (short)0, acc[t], false, false);
        }
    }
}

// ---------------------------------------------------------------------------
// Kernel 1: QKV GEMM.  qkv[n,o] = sum_c x[n,c] * w_qkv[o,c]
// Output stored f16 head-major: qkv_ws[((((b*3+s)*3+br)*4+g)*NPOS+pos)*32+d],
// o = s*384 + br*128 + g*32 + d.
// ---------------------------------------------------------------------------
__global__ __launch_bounds__(256) void qkv_gemm_kernel(
        const _Float16* __restrict__ xh, const _Float16* __restrict__ wqh,
        _Float16* __restrict__ qkv_ws) {
    __shared__ _Float16 Bsh[128 * BSTRIDE];
    const int lane   = threadIdx.x & 31;
    const int waveid = threadIdx.x >> 5;
    const int lane15 = lane & 15;
    const int hsel   = lane >> 4;

    const int ng      = blockIdx.x % (QKVO / 128);   // 9 col groups
    const int mtg     = blockIdx.x / (QKVO / 128);   // 392 row groups of 128
    const int n_base  = ng * 128;
    const int m_base  = (mtg * 8 + waveid) * 16;

    v8f acc[8];
#pragma unroll
    for (int t = 0; t < 8; ++t) acc[t] = (v8f){};
    gemm_panel(xh, wqh, Bsh, m_base, n_base, lane15, hsel, acc);

    // store: tile rows all within one batch image (16 | 3136)
    const int b_img    = m_base / NPOS;
    const int pos_base = m_base % NPOS;
#pragma unroll
    for (int t = 0; t < 8; ++t) {
        const int o  = n_base + t * 16 + lane15;
        const int s  = o / 384;
        const int br = (o / 128) % 3;
        const int g  = (o / 32) % 4;
        const int d  = o % 32;
        const size_t chan = (size_t)(((b_img * 3 + s) * 3 + br) * 4 + g);
#pragma unroll
        for (int i = 0; i < 8; ++i) {
            const int pos = pos_base + hsel * 8 + i;
            qkv_ws[(chan * NPOS + pos) * 32 + d] = (_Float16)acc[t][i];
        }
    }
}

// ---------------------------------------------------------------------------
// Kernel 2: dilated local attention. One thread per (b, branch, head, pos).
// Zero-padded K/V => OOB neighbor logit = 0 (kept in softmax), v = 0.
// ---------------------------------------------------------------------------
__global__ __launch_bounds__(256) void dilate_attn_kernel(
        const _Float16* __restrict__ qkv_ws, _Float16* __restrict__ y_ws) {
    const int t   = blockIdx.x * blockDim.x + threadIdx.x;
    const int pos = t % NPOS;
    const int bg  = t / NPOS;            // 0..191 = ((b*3+br)*4+g)
    const int g   = bg % 4;
    const int br  = (bg / 4) % 3;
    const int b   = bg / 12;
    const int dil = br + 1;
    const int hh  = pos / N_W;
    const int ww  = pos % N_W;

    const size_t qc = (size_t)(((b * 3 + 0) * 3 + br) * 4 + g) * NPOS;
    const size_t kc = (size_t)(((b * 3 + 1) * 3 + br) * 4 + g) * NPOS;
    const size_t vc = (size_t)(((b * 3 + 2) * 3 + br) * 4 + g) * NPOS;

    float q[HDIM];
    {
        const v8h* qp = (const v8h*)(qkv_ws + (qc + pos) * 32);
#pragma unroll
        for (int c8 = 0; c8 < 4; ++c8) {
            v8h qq = qp[c8];
#pragma unroll
            for (int j = 0; j < 8; ++j) q[c8 * 8 + j] = (float)qq[j];
        }
    }

    const float scale = 0.17677669529663687f;  // 32^-0.5
    float logit[9];
    int   npos[9];
#pragma unroll
    for (int ki = 0; ki < 3; ++ki) {
#pragma unroll
        for (int kj = 0; kj < 3; ++kj) {
            const int idx = ki * 3 + kj;
            const int nh = hh + (ki - 1) * dil;
            const int nw = ww + (kj - 1) * dil;
            if (nh >= 0 && nh < N_H && nw >= 0 && nw < N_W) {
                const int np = nh * N_W + nw;
                npos[idx] = np;
                const v8h* kp = (const v8h*)(qkv_ws + (kc + np) * 32);
                float dot = 0.f;
#pragma unroll
                for (int c8 = 0; c8 < 4; ++c8) {
                    v8h kk = kp[c8];
#pragma unroll
                    for (int j = 0; j < 8; ++j) dot += q[c8 * 8 + j] * (float)kk[j];
                }
                logit[idx] = dot * scale;
            } else {
                npos[idx]  = -1;
                logit[idx] = 0.0f;   // dot(q, zero-pad) = 0
            }
        }
    }

    float m = logit[0];
#pragma unroll
    for (int j = 1; j < 9; ++j) m = fmaxf(m, logit[j]);
    float e[9], sum = 0.f;
#pragma unroll
    for (int j = 0; j < 9; ++j) { e[j] = __expf(logit[j] - m); sum += e[j]; }
    const float inv = 1.0f / sum;

    float out[HDIM];
#pragma unroll
    for (int d = 0; d < HDIM; ++d) out[d] = 0.f;
#pragma unroll
    for (int j = 0; j < 9; ++j) {
        if (npos[j] >= 0) {
            const float w = e[j] * inv;
            const v8h* vp = (const v8h*)(qkv_ws + (vc + npos[j]) * 32);
#pragma unroll
            for (int c8 = 0; c8 < 4; ++c8) {
                v8h vv = vp[c8];
#pragma unroll
                for (int jj = 0; jj < 8; ++jj) out[c8 * 8 + jj] += w * (float)vv[jj];
            }
        }
    }

    // y channel order: c = br*128 + g*32 + d
    _Float16* yp = y_ws + (size_t)(b * NPOS + pos) * CIN + br * 128 + g * 32;
#pragma unroll
    for (int d = 0; d < HDIM; ++d) yp[d] = (_Float16)out[d];
}

// ---------------------------------------------------------------------------
// Kernel 3: projection GEMM + bias.  out[n,o] = sum_c y[n,c]*w_proj[o,c] + b[o]
// ---------------------------------------------------------------------------
__global__ __launch_bounds__(256) void proj_gemm_kernel(
        const _Float16* __restrict__ y_ws, const _Float16* __restrict__ wph,
        const float* __restrict__ b_proj, float* __restrict__ out) {
    __shared__ _Float16 Bsh[128 * BSTRIDE];
    const int lane   = threadIdx.x & 31;
    const int waveid = threadIdx.x >> 5;
    const int lane15 = lane & 15;
    const int hsel   = lane >> 4;

    const int ng      = blockIdx.x % (CIN / 128);    // 3 col groups
    const int mtg     = blockIdx.x / (CIN / 128);    // 392 row groups of 128
    const int n_base  = ng * 128;
    const int m_base  = (mtg * 8 + waveid) * 16;

    v8f acc[8];
#pragma unroll
    for (int t = 0; t < 8; ++t) acc[t] = (v8f){};
    gemm_panel(y_ws, wph, Bsh, m_base, n_base, lane15, hsel, acc);

#pragma unroll
    for (int t = 0; t < 8; ++t) {
        const int col    = n_base + t * 16 + lane15;
        const float bias = b_proj[col];
#pragma unroll
        for (int i = 0; i < 8; ++i) {
            const int row = m_base + hsel * 8 + i;
            out[(size_t)row * CIN + col] = acc[t][i] + bias;
        }
    }
}

// ---------------------------------------------------------------------------
extern "C" void kernel_launch(void* const* d_in, const int* in_sizes, int n_in,
                              void* d_out, int out_size, void* d_ws, size_t ws_size,
                              hipStream_t stream) {
    const float* x      = (const float*)d_in[0];   // [50176, 384]
    const float* w_qkv  = (const float*)d_in[1];   // [1152, 384]
    const float* w_proj = (const float*)d_in[2];   // [384, 384]
    const float* b_proj = (const float*)d_in[3];   // [384]
    float* out          = (float*)d_out;           // [50176, 384]

    // workspace layout (all offsets 16B aligned):
    //   xh  f16 [50176,384]   38,535,168 B
    //   wqh f16 [1152,384]       884,736 B
    //   wph f16 [384,384]        294,912 B
    //   qkv f16 [576,3136,32] 115,605,504 B
    //   y   f16 [50176,384]    38,535,168 B
    char* wsp = (char*)d_ws;
    _Float16* xh     = (_Float16*)wsp;                       wsp += (size_t)NROWS * CIN * 2;
    _Float16* wqh    = (_Float16*)wsp;                       wsp += (size_t)QKVO * CIN * 2;
    _Float16* wph    = (_Float16*)wsp;                       wsp += (size_t)CIN * CIN * 2;
    _Float16* qkv_ws = (_Float16*)wsp;                       wsp += (size_t)NROWS * QKVO * 2;
    _Float16* y_ws   = (_Float16*)wsp;

    {   // fp32 -> f16 pre-conversion (keeps GEMM hot loops cvt-free)
        const int n4x = NROWS * CIN / 4;
        cvt_f32_f16_kernel<<<(n4x + 255) / 256, 256, 0, stream>>>(x, xh, n4x);
        const int n4q = QKVO * CIN / 4;
        cvt_f32_f16_kernel<<<(n4q + 255) / 256, 256, 0, stream>>>(w_qkv, wqh, n4q);
        const int n4p = CIN * CIN / 4;
        cvt_f32_f16_kernel<<<(n4p + 255) / 256, 256, 0, stream>>>(w_proj, wph, n4p);
    }
    {   // QKV GEMM: 392 row groups x 9 col groups = 3528 blocks
        qkv_gemm_kernel<<<(NROWS / 128) * (QKVO / 128), 256, 0, stream>>>(xh, wqh, qkv_ws);
    }
    {   // attention: 16*3*4*3136 = 602112 threads
        const int threads = NBATCH * 3 * 4 * NPOS;
        dilate_attn_kernel<<<threads / 256, 256, 0, stream>>>(qkv_ws, y_ws);
    }
    {   // projection GEMM: 392 row groups x 3 col groups = 1176 blocks
        proj_gemm_kernel<<<(NROWS / 128) * (CIN / 128), 256, 0, stream>>>(y_ws, wph, b_proj, out);
    }
}
